// bevdepthLSS_42107859370509
// MI455X (gfx1250) — matrix operand order repaired and checked by reading, not compiled, not run on hardware
//
#include <hip/hip_runtime.h>
#include <hip/hip_bf16.h>
#include <math.h>

typedef __attribute__((ext_vector_type(16))) __bf16 v16bf;
typedef __attribute__((ext_vector_type(8)))  float  v8f;

#define NCAM   6
#define CINCH  512
#define MIDCH  512
#define CCTX   80
#define DDEP   112
#define FHH    16
#define FWW    44
#define NPIX   (FHH * FWW)          // 704
#define MTOT   (NCAM * NPIX)        // 4224
#define KCONV  (MIDCH * 9)          // 4608
#define NPTS   (NCAM * DDEP * NPIX) // 473088
#define BEVN   256
#define BEVCELLS (BEVN * BEVN)      // 65536

#define BMT 128
#define BNT 128
#define BKT 32
#define LDT 40   // padded LDS row stride (bf16 elems): 80 bytes, 16B aligned

// ---------------------------------------------------------------------------
// small helpers
// ---------------------------------------------------------------------------
__device__ inline void inv3(const float* a, float* r) {
    float a0=a[0],a1=a[1],a2=a[2],a3=a[3],a4=a[4],a5=a[5],a6=a[6],a7=a[7],a8=a[8];
    float c0 = a4*a8 - a5*a7;
    float c1 = a5*a6 - a3*a8;
    float c2 = a3*a7 - a4*a6;
    float id = 1.0f / (a0*c0 + a1*c1 + a2*c2);
    r[0]=c0*id; r[1]=(a2*a7-a1*a8)*id; r[2]=(a1*a5-a2*a4)*id;
    r[3]=c1*id; r[4]=(a0*a8-a2*a6)*id; r[5]=(a2*a3-a0*a5)*id;
    r[6]=c2*id; r[7]=(a1*a6-a0*a7)*id; r[8]=(a0*a4-a1*a3)*id;
}

// ---------------------------------------------------------------------------
// zero the output buffer (grid-stride)
// ---------------------------------------------------------------------------
__global__ void zero_k(float* __restrict__ p, long n) {
    long i = (long)blockIdx.x * blockDim.x + threadIdx.x;
    long stride = (long)gridDim.x * blockDim.x;
    for (; i < n; i += stride) p[i] = 0.0f;
}

// ---------------------------------------------------------------------------
// per-camera matrices: geo[cam*21 + 0..8]=inv(post_rots), 9..17=c2l@inv(K), 18..20=trans
// ---------------------------------------------------------------------------
__global__ void prep_mats_k(const float* __restrict__ c2l_rots,
                            const float* __restrict__ c2l_trans,
                            const float* __restrict__ intrins,
                            const float* __restrict__ post_rots,
                            float* __restrict__ geo) {
    int cam = threadIdx.x;
    if (cam >= NCAM) return;
    float ip[9], ii[9];
    inv3(post_rots + cam * 9, ip);
    inv3(intrins  + cam * 9, ii);
    const float* R = c2l_rots + cam * 9;
    float* g = geo + cam * 21;
    for (int i = 0; i < 9; ++i) g[i] = ip[i];
    for (int i = 0; i < 3; ++i)
        for (int j = 0; j < 3; ++j) {
            float s = 0.f;
            for (int k = 0; k < 3; ++k) s += R[i*3+k] * ii[k*3+j];
            g[9 + i*3 + j] = s;
        }
    for (int i = 0; i < 3; ++i) g[18 + i] = c2l_trans[cam*3 + i];
}

// ---------------------------------------------------------------------------
// SE-MLP gates: gates[(branch*6+cam)*512 + c] = sigmoid(exp(relu(red(fc2(relu(fc1(m)))))))
// grid: 12 blocks (cam x branch), 512 threads
// ---------------------------------------------------------------------------
__global__ void mlp_gates_k(const float* __restrict__ mlp_input,
    const float* c_fc1w, const float* c_fc1b, const float* c_fc2w, const float* c_fc2b,
    const float* c_redw, const float* c_redb, const float* c_expw, const float* c_expb,
    const float* d_fc1w, const float* d_fc1b, const float* d_fc2w, const float* d_fc2b,
    const float* d_redw, const float* d_redb, const float* d_expw, const float* d_expb,
    float* __restrict__ gates) {
    __shared__ float buf0[MIDCH];
    __shared__ float buf1[MIDCH];
    int cam = blockIdx.x % NCAM;
    int br  = blockIdx.x / NCAM;
    const float* fc1w = br ? d_fc1w : c_fc1w; const float* fc1b = br ? d_fc1b : c_fc1b;
    const float* fc2w = br ? d_fc2w : c_fc2w; const float* fc2b = br ? d_fc2b : c_fc2b;
    const float* redw = br ? d_redw : c_redw; const float* redb = br ? d_redb : c_redb;
    const float* expw = br ? d_expw : c_expw; const float* expb = br ? d_expb : c_expb;
    int t = threadIdx.x;
    const float* m = mlp_input + cam * 27;
    float a = fc1b[t];
    for (int i = 0; i < 27; ++i) a += fc1w[t*27 + i] * m[i];
    buf0[t] = fmaxf(a, 0.f);
    __syncthreads();
    a = fc2b[t];
    for (int i = 0; i < MIDCH; ++i) a += fc2w[t*MIDCH + i] * buf0[i];
    buf1[t] = a;
    __syncthreads();
    a = redb[t];
    for (int i = 0; i < MIDCH; ++i) a += redw[t*MIDCH + i] * buf1[i];
    buf0[t] = fmaxf(a, 0.f);
    __syncthreads();
    a = expb[t];
    for (int i = 0; i < MIDCH; ++i) a += expw[t*MIDCH + i] * buf0[i];
    gates[(br * NCAM + cam) * MIDCH + t] = 1.0f / (1.0f + expf(-a));
}

// ---------------------------------------------------------------------------
// frustum -> lidar -> BEV flat index per point (or -1 if culled)
// ---------------------------------------------------------------------------
__global__ void geom_flat_k(const float* __restrict__ geo,
                            const float* __restrict__ post_trans,
                            int* __restrict__ flat) {
    int p = blockIdx.x * blockDim.x + threadIdx.x;
    if (p >= NPTS) return;
    int fx = p % FWW; int r1 = p / FWW;
    int fy = r1 % FHH; int r2 = r1 / FHH;
    int dd = r2 % DDEP; int cam = r2 / DDEP;
    float xs  = fx * (703.0f / 43.0f);
    float ys  = fy * (255.0f / 15.0f);
    float dep = 2.0f + 0.5f * (float)dd;
    const float* pt = post_trans + cam * 3;
    float px = xs - pt[0], py = ys - pt[1], pz = dep - pt[2];
    const float* g = geo + cam * 21;
    float qx = g[0]*px + g[1]*py + g[2]*pz;
    float qy = g[3]*px + g[4]*py + g[5]*pz;
    float qz = g[6]*px + g[7]*py + g[8]*pz;
    float ux = qx * qz, uy = qy * qz, uz = qz;
    float rx = g[9]*ux  + g[10]*uy + g[11]*uz + g[18];
    float ry = g[12]*ux + g[13]*uy + g[14]*uz + g[19];
    float rz = g[15]*ux + g[16]*uy + g[17]*uz + g[20];
    int cx = (int)((rx + 51.2f) / 0.4f);
    int cy = (int)((ry + 51.2f) / 0.4f);
    int cz = (int)((rz + 5.0f)  / 8.0f);
    bool kept = (cx >= 0) && (cx < BEVN) && (cy >= 0) && (cy < BEVN) && (cz >= 0) && (cz < 1);
    flat[p] = kept ? (cz * BEVCELLS + cx * BEVN + cy) : -1;
}

// ---------------------------------------------------------------------------
// conv weight [co][ci][3][3] f32 -> Wt[co][(ky*3+kx)*512 + ci] bf16
// ---------------------------------------------------------------------------
__global__ void cvt_w_k(const float* __restrict__ w, __bf16* __restrict__ wt) {
    int tid = blockIdx.x * blockDim.x + threadIdx.x;
    if (tid >= MIDCH * KCONV) return;
    int o  = tid / KCONV;
    int k  = tid % KCONV;
    int kp = k / MIDCH;
    int ci = k % MIDCH;
    int ky = kp / 3, kx = kp % 3;
    wt[tid] = (__bf16)w[(((size_t)o * CINCH + ci) * 3 + ky) * 3 + kx];
}

// contiguous f32 -> bf16 cast
__global__ void cast_k(const float* __restrict__ s, __bf16* __restrict__ d, int n) {
    int tid = blockIdx.x * blockDim.x + threadIdx.x;
    if (tid < n) d[tid] = (__bf16)s[tid];
}

// ---------------------------------------------------------------------------
// im2col: col[m][kp*512 + ci] bf16, 4 channels per thread
// src is NCHW f32 if nchw!=0, else NHWC f32 [img][pix][512]
// ---------------------------------------------------------------------------
__global__ void im2col_k(const float* __restrict__ src, int nchw, __bf16* __restrict__ col) {
    int tid = blockIdx.x * blockDim.x + threadIdx.x;
    const int total = MTOT * 9 * (MIDCH / 4);
    if (tid >= total) return;
    int ci4 = tid % (MIDCH / 4); int r = tid / (MIDCH / 4);
    int kp  = r % 9;             int m = r / 9;
    int img = m / NPIX;          int pix = m % NPIX;
    int y = pix / FWW, x = pix % FWW;
    int ky = kp / 3,   kx = kp % 3;
    int yy = y + ky - 1, xw = x + kx - 1;
    int ci = ci4 * 4;
    __bf16* dst = col + (size_t)m * KCONV + kp * MIDCH + ci;
    if (yy < 0 || yy >= FHH || xw < 0 || xw >= FWW) {
        dst[0] = dst[1] = dst[2] = dst[3] = (__bf16)0.0f;
    } else if (nchw) {
        size_t base = (((size_t)img * CINCH + ci) * FHH + yy) * FWW + xw;
        dst[0] = (__bf16)src[base];
        dst[1] = (__bf16)src[base + (size_t)NPIX];
        dst[2] = (__bf16)src[base + 2 * (size_t)NPIX];
        dst[3] = (__bf16)src[base + 3 * (size_t)NPIX];
    } else {
        const float4 v = *(const float4*)(src + ((size_t)img * NPIX + yy * FWW + xw) * MIDCH + ci);
        dst[0] = (__bf16)v.x; dst[1] = (__bf16)v.y; dst[2] = (__bf16)v.z; dst[3] = (__bf16)v.w;
    }
}

// ---------------------------------------------------------------------------
// h -> gated ctx copy (bf16) and gated dep copy (f32)
// ---------------------------------------------------------------------------
__global__ void gate_split_k(const float* __restrict__ h, const float* __restrict__ gates,
                             __bf16* __restrict__ hc, float* __restrict__ d0) {
    int tid = blockIdx.x * blockDim.x + threadIdx.x;
    if (tid >= MTOT * MIDCH) return;
    int c = tid % MIDCH; int m = tid / MIDCH; int img = m / NPIX;
    float v = h[tid];
    hc[tid] = (__bf16)(v * gates[img * MIDCH + c]);
    d0[tid] = v * gates[(NCAM + img) * MIDCH + c];
}

// ---------------------------------------------------------------------------
// WMMA bf16 GEMM: C[M,N] = A[M,K] * Bt[N,K]^T  (+bias, +res, relu)
// block 256 threads (8 waves), tile 128x128, wave tile 32x64.
// Branch-free staging: OOB rows are clamped (their results never stored);
// register double-buffer hides global load latency behind the WMMAs.
// ---------------------------------------------------------------------------
__device__ inline v16bf frag_a(const __bf16* t, int row0) {
    int lane = threadIdx.x & 31;
    const __bf16* r = t + (row0 + (lane & 15)) * LDT;
    int h = (lane >> 4) * 8;
    v16bf a;
#pragma unroll
    for (int v = 0; v < 8; ++v) {
        int kb = (v < 4 ? 2 * v : 16 + 2 * (v - 4)) + h;
        unsigned u = *(const unsigned*)(r + kb);
        a[2 * v]     = ((const __bf16*)&u)[0];
        a[2 * v + 1] = ((const __bf16*)&u)[1];
    }
    return a;
}

__device__ inline v16bf frag_b(const __bf16* t, int col0) {
    int lane = threadIdx.x & 31;
    const __bf16* r = t + (col0 + (lane & 15)) * LDT;
    int h = (lane >> 4) * 8;
    v16bf b;
#pragma unroll
    for (int v = 0; v < 8; ++v) {
        int kb = (v < 4 ? 2 * v : 16 + 2 * (v - 4)) + h;
        unsigned u = *(const unsigned*)(r + kb);
        b[2 * v]     = ((const __bf16*)&u)[0];
        b[2 * v + 1] = ((const __bf16*)&u)[1];
    }
    return b;
}

__launch_bounds__(256)
__global__ void gemm_k(const __bf16* __restrict__ A, const __bf16* __restrict__ Bt,
                       float* __restrict__ C, const float* __restrict__ bias,
                       const float* __restrict__ res, int M, int N, int K, int relu) {
    __shared__ __bf16 As[BMT * LDT];
    __shared__ __bf16 Bs[BNT * LDT];
    const int m0 = blockIdx.x * BMT;
    const int n0 = blockIdx.y * BNT;
    const int t = threadIdx.x;
    const int wave = t >> 5, lane = t & 31;
    const int wm = (wave & 3) * 32;
    const int wn = (wave >> 2) * 64;
    const int lrow  = t >> 1;
    const int lhalf = (t & 1) * 16;

    const v8f vzero = {0.f, 0.f, 0.f, 0.f, 0.f, 0.f, 0.f, 0.f};
    v8f acc[2][4];
#pragma unroll
    for (int i = 0; i < 2; ++i)
#pragma unroll
        for (int j = 0; j < 4; ++j) acc[i][j] = vzero;

    // clamp OOB rows to a valid row: their products are discarded at store time
    const int gmA = min(m0 + lrow, M - 1);
    const int gnB = min(n0 + lrow, N - 1);
    const __bf16* pA = A  + (size_t)gmA * K + lhalf;
    const __bf16* pB = Bt + (size_t)gnB * K + lhalf;
    __bf16* sA = As + lrow * LDT + lhalf;
    __bf16* sB = Bs + lrow * LDT + lhalf;

    // preload first K-slice into registers
    uint4 ra0 = *(const uint4*)(pA);
    uint4 ra1 = *(const uint4*)(pA + 8);
    uint4 rb0 = *(const uint4*)(pB);
    uint4 rb1 = *(const uint4*)(pB + 8);

    for (int k0 = 0; k0 < K; k0 += BKT) {
        *(uint4*)(sA)     = ra0;
        *(uint4*)(sA + 8) = ra1;
        *(uint4*)(sB)     = rb0;
        *(uint4*)(sB + 8) = rb1;
        __syncthreads();
        const int kn = k0 + BKT;
        if (kn < K) {   // prefetch next slice while WMMAs consume current LDS tile
            ra0 = *(const uint4*)(pA + kn);
            ra1 = *(const uint4*)(pA + kn + 8);
            rb0 = *(const uint4*)(pB + kn);
            rb1 = *(const uint4*)(pB + kn + 8);
        }
        v16bf af0 = frag_a(As, wm);
        v16bf af1 = frag_a(As, wm + 16);
        v16bf bf0 = frag_b(Bs, wn);
        v16bf bf1 = frag_b(Bs, wn + 16);
        v16bf bf2 = frag_b(Bs, wn + 32);
        v16bf bf3 = frag_b(Bs, wn + 48);
        acc[0][0] = __builtin_amdgcn_wmma_f32_16x16x32_bf16(false, af0, false, bf0, (short)0, acc[0][0], false, false);
        acc[0][1] = __builtin_amdgcn_wmma_f32_16x16x32_bf16(false, af0, false, bf1, (short)0, acc[0][1], false, false);
        acc[0][2] = __builtin_amdgcn_wmma_f32_16x16x32_bf16(false, af0, false, bf2, (short)0, acc[0][2], false, false);
        acc[0][3] = __builtin_amdgcn_wmma_f32_16x16x32_bf16(false, af0, false, bf3, (short)0, acc[0][3], false, false);
        acc[1][0] = __builtin_amdgcn_wmma_f32_16x16x32_bf16(false, af1, false, bf0, (short)0, acc[1][0], false, false);
        acc[1][1] = __builtin_amdgcn_wmma_f32_16x16x32_bf16(false, af1, false, bf1, (short)0, acc[1][1], false, false);
        acc[1][2] = __builtin_amdgcn_wmma_f32_16x16x32_bf16(false, af1, false, bf2, (short)0, acc[1][2], false, false);
        acc[1][3] = __builtin_amdgcn_wmma_f32_16x16x32_bf16(false, af1, false, bf3, (short)0, acc[1][3], false, false);
        __syncthreads();
    }

    const int cn = lane & 15;
    const int ch = (lane >> 4) * 8;
#pragma unroll
    for (int i = 0; i < 2; ++i) {
#pragma unroll
        for (int j = 0; j < 4; ++j) {
            int gn = n0 + wn + j * 16 + cn;
            if (gn >= N) continue;
            float bv = bias ? bias[gn] : 0.f;
#pragma unroll
            for (int rr = 0; rr < 8; ++rr) {
                int gm = m0 + wm + i * 16 + ch + rr;
                if (gm >= M) continue;
                float v = acc[i][j][rr] + bv;
                if (res)  v += res[(size_t)gm * N + gn];
                if (relu) v = fmaxf(v, 0.f);
                C[(size_t)gm * N + gn] = v;
            }
        }
    }
}

// ---------------------------------------------------------------------------
// per-row softmax over 112 depth bins (one wave32 per row, in place)
// ---------------------------------------------------------------------------
__global__ void softmax_k(float* __restrict__ logits) {
    int row = blockIdx.x * (blockDim.x >> 5) + (threadIdx.x >> 5);
    if (row >= MTOT) return;
    int lane = threadIdx.x & 31;
    float* p = logits + (size_t)row * DDEP;
    float vals[4];
    float mx = -1e30f;
#pragma unroll
    for (int i = 0; i < 4; ++i) {
        int c = lane + 32 * i;
        vals[i] = (c < DDEP) ? p[c] : -1e30f;
        mx = fmaxf(mx, vals[i]);
    }
    for (int off = 16; off; off >>= 1) mx = fmaxf(mx, __shfl_xor(mx, off, 32));
    float s = 0.f;
#pragma unroll
    for (int i = 0; i < 4; ++i) {
        int c = lane + 32 * i;
        vals[i] = (c < DDEP) ? expf(vals[i] - mx) : 0.f;
        s += vals[i];
    }
    for (int off = 16; off; off >>= 1) s += __shfl_xor(s, off, 32);
    float inv = 1.f / s;
#pragma unroll
    for (int i = 0; i < 4; ++i) {
        int c = lane + 32 * i;
        if (c < DDEP) p[c] = vals[i] * inv;
    }
}

// ---------------------------------------------------------------------------
// depth x context outer product + BEV scatter-add (out layout [80][256][256])
// ---------------------------------------------------------------------------
__global__ void scatter_k(const float* __restrict__ depth, const float* __restrict__ ctx,
                          const int* __restrict__ flat, float* __restrict__ out) {
    int p = blockIdx.x * blockDim.x + threadIdx.x;
    if (p >= NPTS) return;
    int f = flat[p];
    if (f < 0) return;
    int fx = p % FWW; int r1 = p / FWW;
    int fy = r1 % FHH; int r2 = r1 / FHH;
    int dd = r2 % DDEP; int cam = r2 / DDEP;
    int m = cam * NPIX + fy * FWW + fx;
    float w = depth[(size_t)m * DDEP + dd];
    const float* c = ctx + (size_t)m * CCTX;
#pragma unroll 4
    for (int ch = 0; ch < CCTX; ++ch)
        atomicAdd(out + (size_t)ch * BEVCELLS + f, w * c[ch]);
}

// ---------------------------------------------------------------------------
// host launcher
// ---------------------------------------------------------------------------
extern "C" void kernel_launch(void* const* d_in, const int* in_sizes, int n_in,
                              void* d_out, int out_size, void* d_ws, size_t ws_size,
                              hipStream_t stream) {
    (void)in_sizes; (void)n_in; (void)out_size; (void)ws_size;
    const float* x          = (const float*)d_in[0];
    const float* mlp_input  = (const float*)d_in[1];
    const float* c2l_rots   = (const float*)d_in[2];
    const float* c2l_trans  = (const float*)d_in[3];
    const float* intrins    = (const float*)d_in[4];
    const float* post_rots  = (const float*)d_in[5];
    const float* post_trans = (const float*)d_in[6];
    const float* reduce_w   = (const float*)d_in[7];
    const float* reduce_b   = (const float*)d_in[8];
    const float* ctx_fc1_w  = (const float*)d_in[9];
    const float* ctx_fc1_b  = (const float*)d_in[10];
    const float* ctx_fc2_w  = (const float*)d_in[11];
    const float* ctx_fc2_b  = (const float*)d_in[12];
    const float* ctx_red_w  = (const float*)d_in[13];
    const float* ctx_red_b  = (const float*)d_in[14];
    const float* ctx_exp_w  = (const float*)d_in[15];
    const float* ctx_exp_b  = (const float*)d_in[16];
    const float* ctx_out_w  = (const float*)d_in[17];
    const float* ctx_out_b  = (const float*)d_in[18];
    const float* dep_fc1_w  = (const float*)d_in[19];
    const float* dep_fc1_b  = (const float*)d_in[20];
    const float* dep_fc2_w  = (const float*)d_in[21];
    const float* dep_fc2_b  = (const float*)d_in[22];
    const float* dep_red_w  = (const float*)d_in[23];
    const float* dep_red_b  = (const float*)d_in[24];
    const float* dep_exp_w  = (const float*)d_in[25];
    const float* dep_exp_b  = (const float*)d_in[26];
    const float* blk_w1     = (const float*)d_in[27];
    const float* blk_w2     = (const float*)d_in[28];
    const float* dep_out_w  = (const float*)d_in[29];
    const float* dep_out_b  = (const float*)d_in[30];

    char* wp = (char*)d_ws;
    auto alloc = [&](size_t bytes) {
        char* r = wp;
        wp += (bytes + 255) & ~(size_t)255;
        return r;
    };
    __bf16* col    = (__bf16*)alloc((size_t)MTOT * KCONV * 2);
    __bf16* wconv  = (__bf16*)alloc((size_t)MIDCH * KCONV * 2);
    __bf16* wlin   = (__bf16*)alloc((size_t)DDEP * MIDCH * 2);
    float*  hbuf   = (float*) alloc((size_t)MTOT * MIDCH * 4);
    float*  dA     = (float*) alloc((size_t)MTOT * MIDCH * 4);
    float*  dB     = (float*) alloc((size_t)MTOT * MIDCH * 4);
    float*  obuf   = (float*) alloc((size_t)MTOT * MIDCH * 4);
    __bf16* hcb    = (__bf16*)alloc((size_t)MTOT * MIDCH * 2);
    __bf16* dbf    = (__bf16*)alloc((size_t)MTOT * MIDCH * 2);
    float*  ctxb   = (float*) alloc((size_t)MTOT * CCTX * 4);
    float*  logits = (float*) alloc((size_t)MTOT * DDEP * 4);
    float*  gates  = (float*) alloc((size_t)2 * NCAM * MIDCH * 4);
    float*  geo    = (float*) alloc((size_t)NCAM * 21 * 4);
    int*    flat   = (int*)   alloc((size_t)NPTS * 4);

    const int T = 256;
    const int g_geom   = (NPTS + T - 1) / T;
    const int g_cvtw   = (MIDCH * KCONV + T - 1) / T;
    const int g_im2col = (MTOT * 9 * (MIDCH / 4) + T - 1) / T;
    const int g_act    = (MTOT * MIDCH + T - 1) / T;

    // small prep
    zero_k<<<2048, T, 0, stream>>>((float*)d_out, (long)CCTX * BEVCELLS);
    prep_mats_k<<<1, 32, 0, stream>>>(c2l_rots, c2l_trans, intrins, post_rots, geo);
    mlp_gates_k<<<12, 512, 0, stream>>>(mlp_input,
        ctx_fc1_w, ctx_fc1_b, ctx_fc2_w, ctx_fc2_b, ctx_red_w, ctx_red_b, ctx_exp_w, ctx_exp_b,
        dep_fc1_w, dep_fc1_b, dep_fc2_w, dep_fc2_b, dep_red_w, dep_red_b, dep_exp_w, dep_exp_b,
        gates);
    geom_flat_k<<<g_geom, T, 0, stream>>>(geo, post_trans, flat);

    // reduce conv: h = relu(conv3x3(x) + b)
    cvt_w_k<<<g_cvtw, T, 0, stream>>>(reduce_w, wconv);
    im2col_k<<<g_im2col, T, 0, stream>>>(x, 1, col);
    gemm_k<<<dim3(MTOT / BMT, MIDCH / BNT), T, 0, stream>>>(
        col, wconv, hbuf, reduce_b, nullptr, MTOT, MIDCH, KCONV, 1);

    // SE gating -> ctx bf16 copy + dep f32 copy
    gate_split_k<<<g_act, T, 0, stream>>>(hbuf, gates, hcb, dA);

    // context head: [4224,512] x [512,80] + bias
    cast_k<<<(CCTX * MIDCH + T - 1) / T, T, 0, stream>>>(ctx_out_w, wlin, CCTX * MIDCH);
    gemm_k<<<dim3(MTOT / BMT, 1), T, 0, stream>>>(
        hcb, wlin, ctxb, ctx_out_b, nullptr, MTOT, CCTX, MIDCH, 0);

    // 3 BasicBlocks: d = relu(conv2(relu(conv1(d))) + d)
    float* dcur = dA;
    float* dnext = dB;
    for (int i = 0; i < 3; ++i) {
        const size_t woff = (size_t)i * MIDCH * MIDCH * 9;
        cvt_w_k<<<g_cvtw, T, 0, stream>>>(blk_w1 + woff, wconv);
        im2col_k<<<g_im2col, T, 0, stream>>>(dcur, 0, col);
        gemm_k<<<dim3(MTOT / BMT, MIDCH / BNT), T, 0, stream>>>(
            col, wconv, obuf, nullptr, nullptr, MTOT, MIDCH, KCONV, 1);
        cvt_w_k<<<g_cvtw, T, 0, stream>>>(blk_w2 + woff, wconv);
        im2col_k<<<g_im2col, T, 0, stream>>>(obuf, 0, col);
        gemm_k<<<dim3(MTOT / BMT, MIDCH / BNT), T, 0, stream>>>(
            col, wconv, dnext, nullptr, dcur, MTOT, MIDCH, KCONV, 1);
        float* tmp = dcur; dcur = dnext; dnext = tmp;
    }

    // depth head: logits = d @ dep_out_w^T + b, then softmax over 112
    cast_k<<<g_act, T, 0, stream>>>(dcur, dbf, MTOT * MIDCH);
    cast_k<<<(DDEP * MIDCH + T - 1) / T, T, 0, stream>>>(dep_out_w, wlin, DDEP * MIDCH);
    gemm_k<<<dim3(MTOT / BMT, 1), T, 0, stream>>>(
        dbf, wlin, logits, dep_out_b, nullptr, MTOT, DDEP, MIDCH, 0);
    softmax_k<<<(MTOT + 3) / 4, 128, 0, stream>>>(logits);

    // lift-splat scatter into BEV
    scatter_k<<<g_geom, T, 0, stream>>>(logits, ctxb, flat, (float*)d_out);
}
